// InteractionBlock_63582695850652
// MI455X (gfx1250) — matrix operand order repaired and checked by reading, not compiled, not run on hardware
//
#include <hip/hip_runtime.h>
#include <cstdint>
#include <cstddef>

// ---------------------------------------------------------------------------
// Types for CDNA5 WMMA (wave32, 16x16x32 bf16 -> f32)
// ---------------------------------------------------------------------------
typedef __attribute__((ext_vector_type(16))) __bf16        v16bf;
typedef __attribute__((ext_vector_type(8)))  float         v8f;
typedef __attribute__((ext_vector_type(4)))  unsigned int  u32x4;

union Frag {
    v16bf          v;
    u32x4          q[2];
    unsigned short u[16];
};

constexpr int kE   = 128;     // edge message dim
constexpr int kNR  = 32;      // n_radial
constexpr int kNSR = 224;     // n_spherical * n_radial
constexpr int kNB  = 8;       // n_bilinear
constexpr int kNE  = 100000;  // edges
constexpr int kNT  = 200000;  // triplets
constexpr int kEdgeTiles = kNE / 16;  // 6250
constexpr int kTriTiles  = kNT / 16;  // 12500

__device__ __forceinline__ unsigned short f32_to_bf16(float f) {
    unsigned int u = __float_as_uint(f);
    u += 0x7fffu + ((u >> 16) & 1u);          // round-to-nearest-even
    return (unsigned short)(u >> 16);
}
__device__ __forceinline__ float bf16_to_f32(unsigned short h) {
    return __uint_as_float(((unsigned int)h) << 16);
}
__device__ __forceinline__ float silu_f(float v) {
    return v / (1.0f + __expf(-v));
}

// Async DMA: copy 16 bytes global -> LDS without VGPR staging (ASYNCcnt).
__device__ __forceinline__ void async_g2l_b128(unsigned int lds_byte_addr,
                                               unsigned long long gaddr) {
    asm volatile("global_load_async_to_lds_b128 %0, %1, off"
                 :: "v"(lds_byte_addr), "v"(gaddr)
                 : "memory");
}
__device__ __forceinline__ void wait_async0() {
    asm volatile("s_wait_asynccnt 0" ::: "memory");
}

// ---------------------------------------------------------------------------
// 16xN GEMM tile via v_wmma_f32_16x16x32_bf16.
//   A: 16 x kdim bf16, row-major, base row pointer a_base, row stride astride
//   W: N(=128 here, 8 n-tiles) x kdim bf16 row-major (i.e. B[k][n] = W[n][k])
//   acc[n][r] -> (M = r + 8*hi, Ncol = n*16 + row)
// A-frag (16x32 bf16): lanes 0-15 K={0..7,16..23}, lanes 16-31 K={8..15,24..31}
// B-frag (32x16 bf16): lanes 0-15 K={0..7,16..23} of col (lane&15); lanes16-31 K={8..15,24..31}
// ---------------------------------------------------------------------------
__device__ __forceinline__ void gemm16(const unsigned short* a_base, int astride,
                                       const unsigned short* w, int kdim,
                                       v8f acc[8], int row, int hi) {
#pragma unroll
    for (int n = 0; n < 8; n++)
#pragma unroll
        for (int r = 0; r < 8; r++) acc[n][r] = 0.0f;
    const int nk = kdim >> 5;
#pragma unroll
    for (int kc = 0; kc < nk; kc++) {
        Frag a;
        const unsigned short* ap = a_base + row * astride + kc * 32 + hi * 8;
        a.q[0] = *(const u32x4*)ap;
        a.q[1] = *(const u32x4*)(ap + 16);
#pragma unroll
        for (int n = 0; n < 8; n++) {
            Frag b;
            const unsigned short* bp = w + (size_t)(n * 16 + row) * kdim + kc * 32 + hi * 8;
            b.q[0] = *(const u32x4*)bp;
            b.q[1] = *(const u32x4*)(bp + 16);
            acc[n] = __builtin_amdgcn_wmma_f32_16x16x32_bf16(
                false, a.v, false, b.v, (short)0, acc[n], false, false);
        }
    }
}

// ---------------------------------------------------------------------------
// Kernel: f32 -> bf16 conversion
// ---------------------------------------------------------------------------
__global__ __launch_bounds__(256) void cvt_f32_bf16_kernel(
    const float* __restrict__ s, unsigned short* __restrict__ d, int n) {
    int i = blockIdx.x * blockDim.x + threadIdx.x;
    if (i < n) d[i] = f32_to_bf16(s[i]);
}

// ---------------------------------------------------------------------------
// Kernel 1: edge projections
//   rbf_p = rbf @ w_rbf^T ; x_kj = silu(x@w_kj^T+b)*rbf_p -> bf16
//   x_ji  = silu(x@w_ji^T+b) -> f32 written into agg (segment-sum init)
// One 16-edge tile per wave.
// ---------------------------------------------------------------------------
__global__ __launch_bounds__(256) void edge_proj_kernel(
    const unsigned short* __restrict__ x_bf, const unsigned short* __restrict__ rbf_bf,
    const unsigned short* __restrict__ w_ji_bf, const float* __restrict__ b_ji,
    const unsigned short* __restrict__ w_kj_bf, const float* __restrict__ b_kj,
    const unsigned short* __restrict__ w_rbf_bf,
    float* __restrict__ agg, unsigned short* __restrict__ xkj_bf) {
    int wave = threadIdx.x >> 5, lane = threadIdx.x & 31;
    int row = lane & 15, hi = lane >> 4;
    int tile = blockIdx.x * 8 + wave;
    if (tile >= kEdgeTiles) return;
    int e0 = tile * 16;

    v8f accr[8], acc[8];
    // rbf_p (K = 32)
    gemm16(rbf_bf + (size_t)e0 * kNR, kNR, w_rbf_bf, kNR, accr, row, hi);
    // x_kj (K = 128)
    gemm16(x_bf + (size_t)e0 * kE, kE, w_kj_bf, kE, acc, row, hi);
#pragma unroll
    for (int n = 0; n < 8; n++) {
        int N = n * 16 + row;
        float bias = b_kj[N];
#pragma unroll
        for (int r = 0; r < 8; r++) {
            int M = r + hi * 8;
            float v = silu_f(acc[n][r] + bias) * accr[n][r];
            xkj_bf[(size_t)(e0 + M) * kE + N] = f32_to_bf16(v);
        }
    }
    // x_ji (K = 128), write f32 into agg
    gemm16(x_bf + (size_t)e0 * kE, kE, w_ji_bf, kE, acc, row, hi);
#pragma unroll
    for (int n = 0; n < 8; n++) {
        int N = n * 16 + row;
        float bias = b_ji[N];
#pragma unroll
        for (int r = 0; r < 8; r++) {
            int M = r + hi * 8;
            agg[(size_t)(e0 + M) * kE + N] = silu_f(acc[n][r] + bias);
        }
    }
}

// ---------------------------------------------------------------------------
// Kernel 2: sbf_p = sbf @ w_sbf^T  (NT x 224) @ (224 x 8)  — VALU, w in LDS
// ---------------------------------------------------------------------------
__global__ __launch_bounds__(256) void sbf_proj_kernel(
    const float* __restrict__ sbf, const float* __restrict__ w_sbf,
    float* __restrict__ sbf_p) {
    __shared__ float ws[kNB * kNSR];
    for (int i = threadIdx.x; i < kNB * kNSR; i += blockDim.x) ws[i] = w_sbf[i];
    __syncthreads();
    int t = blockIdx.x * blockDim.x + threadIdx.x;
    if (t >= kNT) return;
    const float4* rowp = (const float4*)(sbf + (size_t)t * kNSR);
    __builtin_prefetch(rowp, 0, 1);  // global_prefetch_b8: pull row toward L2/L0
    float acc[kNB];
#pragma unroll
    for (int b = 0; b < kNB; b++) acc[b] = 0.0f;
#pragma unroll 7
    for (int k4 = 0; k4 < kNSR / 4; k4++) {
        float4 v = rowp[k4];
        int k = k4 * 4;
#pragma unroll
        for (int b = 0; b < kNB; b++) {
            acc[b] += v.x * ws[b * kNSR + k + 0] + v.y * ws[b * kNSR + k + 1] +
                      v.z * ws[b * kNSR + k + 2] + v.w * ws[b * kNSR + k + 3];
        }
    }
#pragma unroll
    for (int b = 0; b < kNB; b++) sbf_p[(size_t)t * kNB + b] = acc[b];
}

// ---------------------------------------------------------------------------
// Kernel 3: bilinear + scatter-add.
//   m[t,o] = sum_{b,e} sbf_p[t,b]*xk[t,e]*w_bil[o,b,e]
// Folded as one K=1024 GEMM; grid.y in {0,1} selects a 4-b half (K'=512)
// whose weights (128 KB bf16) are DMAed into LDS with
// global_load_async_to_lds_b128 (ASYNCcnt), as is the gathered per-wave
// 16x128 xk tile. A' built on the fly: per 32-wide K-chunk the sbf_p scale
// is constant. Results atomically added into agg.
// 8 waves/WG, one 16-triplet tile per wave. LDS = 128KB + 8*4KB = 160KB.
// ---------------------------------------------------------------------------
__global__ __launch_bounds__(256) void bilinear_kernel(
    const unsigned short* __restrict__ xkj_bf, const unsigned short* __restrict__ wbil_bf,
    const float* __restrict__ sbf_p, const int* __restrict__ idx_kj,
    const int* __restrict__ idx_ji, float* __restrict__ agg) {
    extern __shared__ char smem[];
    unsigned short* wlds = (unsigned short*)smem;                       // 128 x 512 bf16
    unsigned short* xk   = (unsigned short*)(smem + 128 * 512 * 2) +
                           (threadIdx.x >> 5) * (16 * kE);              // per-wave 16x128
    int wave = threadIdx.x >> 5, lane = threadIdx.x & 31;
    int row = lane & 15, hi = lane >> 4;

    // async DMA weight half: wlds[o][k'] = w_bil[o][b0*128 + k'], 512B/thread
    {
        int o = threadIdx.x >> 1, half = threadIdx.x & 1;
        unsigned long long gsrc = (unsigned long long)(size_t)(
            wbil_bf + (size_t)o * 1024 + (size_t)blockIdx.y * 512 + half * 256);
        unsigned int ldst = (unsigned int)(size_t)(wlds + o * 512 + half * 256);
#pragma unroll
        for (int i = 0; i < 32; i++)
            async_g2l_b128(ldst + i * 16, gsrc + i * 16);
    }
    int tile = blockIdx.x * 8 + wave;
    bool active = tile < kTriTiles;

    if (active) {  // async DMA gather of 16 xk rows into this wave's LDS tile
        int r2 = lane >> 1, half = lane & 1;
        int eidx = idx_kj[tile * 16 + r2];
        unsigned long long gsrc = (unsigned long long)(size_t)(
            xkj_bf + (size_t)eidx * kE + half * 64);
        unsigned int ldst = (unsigned int)(size_t)(xk + r2 * kE + half * 64);
#pragma unroll
        for (int i = 0; i < 8; i++)
            async_g2l_b128(ldst + i * 16, gsrc + i * 16);
    }
    wait_async0();        // drain this wave's ASYNCcnt
    __syncthreads();      // cross-wave visibility of wlds
    if (!active) return;

    float sv[4];
#pragma unroll
    for (int bl = 0; bl < 4; bl++)
        sv[bl] = sbf_p[(size_t)(tile * 16 + row) * kNB + blockIdx.y * 4 + bl];

    v8f acc[8];
#pragma unroll
    for (int n = 0; n < 8; n++)
#pragma unroll
        for (int r = 0; r < 8; r++) acc[n][r] = 0.0f;

#pragma unroll
    for (int c = 0; c < 16; c++) {            // 16 K-chunks of 32 (K'=512)
        float s = sv[c >> 2];
        const unsigned short* ap = xk + row * kE + (c & 3) * 32 + hi * 8;
        Frag a;
#pragma unroll
        for (int j = 0; j < 8; j++) a.u[j] = f32_to_bf16(bf16_to_f32(ap[j]) * s);
#pragma unroll
        for (int j = 0; j < 8; j++) a.u[8 + j] = f32_to_bf16(bf16_to_f32(ap[16 + j]) * s);
#pragma unroll
        for (int n = 0; n < 8; n++) {
            Frag b;
            const unsigned short* bp = wlds + (size_t)(n * 16 + row) * 512 + c * 32 + hi * 8;
            b.q[0] = *(const u32x4*)bp;
            b.q[1] = *(const u32x4*)(bp + 16);
            acc[n] = __builtin_amdgcn_wmma_f32_16x16x32_bf16(
                false, a.v, false, b.v, (short)0, acc[n], false, false);
        }
    }
    // scatter-add via segment indices
    int ji[8];
#pragma unroll
    for (int r = 0; r < 8; r++) ji[r] = idx_ji[tile * 16 + r + hi * 8];
#pragma unroll
    for (int n = 0; n < 8; n++) {
        int N = n * 16 + row;
#pragma unroll
        for (int r = 0; r < 8; r++)
            atomicAdd(agg + (size_t)ji[r] * kE + N, acc[n][r]);
    }
}

// ---------------------------------------------------------------------------
// Kernel 4: fused epilogue (rb1, bs+skip(x), rb2, rb3) — 7 GEMMs per tile,
// activations ping-pong through per-wave LDS: bf16 GEMM-input tile + f32 skip.
// ---------------------------------------------------------------------------
__device__ __forceinline__ void res_block(unsigned short* hb, float* sk,
                                          const unsigned short* w1, const float* bias1,
                                          const unsigned short* w2, const float* bias2,
                                          int row, int hi, bool active) {
    v8f acc[8];
    if (active) gemm16(hb, kE, w1, kE, acc, row, hi);
    __syncthreads();
    if (active) {
#pragma unroll
        for (int n = 0; n < 8; n++) {
            int N = n * 16 + row;
            float bn = bias1[N];
#pragma unroll
            for (int r = 0; r < 8; r++) {
                int M = r + hi * 8;
                hb[M * kE + N] = f32_to_bf16(silu_f(acc[n][r] + bn));
            }
        }
    }
    __syncthreads();
    if (active) gemm16(hb, kE, w2, kE, acc, row, hi);
    __syncthreads();
    if (active) {
#pragma unroll
        for (int n = 0; n < 8; n++) {
            int N = n * 16 + row;
            float bn = bias2[N];
#pragma unroll
            for (int r = 0; r < 8; r++) {
                int M = r + hi * 8, idx = M * kE + N;
                float v = sk[idx] + silu_f(acc[n][r] + bn);
                sk[idx] = v;
                hb[idx] = f32_to_bf16(v);
            }
        }
    }
    __syncthreads();
}

__global__ __launch_bounds__(256) void epilogue_kernel(
    const float* __restrict__ agg, const float* __restrict__ x,
    const unsigned short* w1a, const float* b1a, const unsigned short* w1b, const float* b1b,
    const unsigned short* wbs, const float* bbs,
    const unsigned short* w2a, const float* b2a, const unsigned short* w2b, const float* b2b,
    const unsigned short* w3a, const float* b3a, const unsigned short* w3b, const float* b3b,
    float* __restrict__ out) {
    extern __shared__ char smem[];
    int wave = threadIdx.x >> 5, lane = threadIdx.x & 31;
    int row = lane & 15, hi = lane >> 4;
    unsigned short* hb = (unsigned short*)smem + wave * 16 * kE;             // bf16 tile
    float* sk = (float*)(smem + 8 * 16 * kE * 2) + wave * 16 * kE;           // f32 skip
    int tile = blockIdx.x * 8 + wave;
    bool active = tile < kEdgeTiles;

    if (active) {  // load agg tile -> skip (f32) + GEMM input (bf16)
        int r2 = lane >> 1, half = lane & 1;
        const float4* s = (const float4*)(agg + (size_t)(tile * 16 + r2) * kE + half * 64);
        float* skp = sk + r2 * kE + half * 64;
        unsigned short* hp = hb + r2 * kE + half * 64;
#pragma unroll
        for (int i = 0; i < 16; i++) {
            float4 v = s[i];
            ((float4*)skp)[i] = v;
            hp[i * 4 + 0] = f32_to_bf16(v.x);
            hp[i * 4 + 1] = f32_to_bf16(v.y);
            hp[i * 4 + 2] = f32_to_bf16(v.z);
            hp[i * 4 + 3] = f32_to_bf16(v.w);
        }
    }
    __syncthreads();

    res_block(hb, sk, w1a, b1a, w1b, b1b, row, hi, active);

    {  // bs layer: h = silu(h@bs_w^T + b) + x
        v8f acc[8];
        if (active) gemm16(hb, kE, wbs, kE, acc, row, hi);
        __syncthreads();
        if (active) {
#pragma unroll
            for (int n = 0; n < 8; n++) {
                int N = n * 16 + row;
                float bn = bbs[N];
#pragma unroll
                for (int r = 0; r < 8; r++) {
                    int M = r + hi * 8, idx = M * kE + N;
                    float v = silu_f(acc[n][r] + bn) + x[(size_t)(tile * 16 + M) * kE + N];
                    sk[idx] = v;
                    hb[idx] = f32_to_bf16(v);
                }
            }
        }
        __syncthreads();
    }

    res_block(hb, sk, w2a, b2a, w2b, b2b, row, hi, active);
    res_block(hb, sk, w3a, b3a, w3b, b3b, row, hi, active);

    if (active) {  // final result is in sk
        int r2 = lane >> 1, half = lane & 1;
        const float4* s = (const float4*)(sk + r2 * kE + half * 64);
        float4* d = (float4*)(out + (size_t)(tile * 16 + r2) * kE + half * 64);
#pragma unroll
        for (int i = 0; i < 16; i++) d[i] = s[i];
    }
}

// ---------------------------------------------------------------------------
// Host launcher
// ---------------------------------------------------------------------------
extern "C" void kernel_launch(void* const* d_in, const int* in_sizes, int n_in,
                              void* d_out, int out_size, void* d_ws, size_t ws_size,
                              hipStream_t stream) {
    (void)in_sizes; (void)n_in; (void)out_size;
    const float* x      = (const float*)d_in[0];
    const float* rbf    = (const float*)d_in[1];
    const float* sbf    = (const float*)d_in[2];
    const int*   idx_kj = (const int*)d_in[3];
    const int*   idx_ji = (const int*)d_in[4];
    const float* w_rbf  = (const float*)d_in[5];
    const float* w_sbf  = (const float*)d_in[6];
    const float* w_kj   = (const float*)d_in[7];
    const float* b_kj   = (const float*)d_in[8];
    const float* w_ji   = (const float*)d_in[9];
    const float* b_ji   = (const float*)d_in[10];
    const float* w_bil  = (const float*)d_in[11];
    const float* rb1w1  = (const float*)d_in[12]; const float* rb1b1 = (const float*)d_in[13];
    const float* rb1w2  = (const float*)d_in[14]; const float* rb1b2 = (const float*)d_in[15];
    const float* bsw    = (const float*)d_in[16]; const float* bsb   = (const float*)d_in[17];
    const float* rb2w1  = (const float*)d_in[18]; const float* rb2b1 = (const float*)d_in[19];
    const float* rb2w2  = (const float*)d_in[20]; const float* rb2b2 = (const float*)d_in[21];
    const float* rb3w1  = (const float*)d_in[22]; const float* rb3b1 = (const float*)d_in[23];
    const float* rb3w2  = (const float*)d_in[24]; const float* rb3b2 = (const float*)d_in[25];

    char* ws = (char*)d_ws;
    size_t off = 0;
    auto alloc = [&](size_t bytes) -> void* {
        void* p = ws + off;
        off += (bytes + 255) & ~(size_t)255;
        return p;
    };
    unsigned short* x_bf     = (unsigned short*)alloc((size_t)kNE * kE * 2);
    unsigned short* rbf_bf   = (unsigned short*)alloc((size_t)kNE * kNR * 2);
    unsigned short* xkj_bf   = (unsigned short*)alloc((size_t)kNE * kE * 2);
    float*          sbf_p    = (float*)alloc((size_t)kNT * kNB * 4);
    float*          agg      = (float*)alloc((size_t)kNE * kE * 4);
    unsigned short* w_rbf_bf = (unsigned short*)alloc((size_t)kE * kNR * 2);
    unsigned short* w_kj_bf  = (unsigned short*)alloc((size_t)kE * kE * 2);
    unsigned short* w_ji_bf  = (unsigned short*)alloc((size_t)kE * kE * 2);
    unsigned short* wbil_bf  = (unsigned short*)alloc((size_t)kE * kNB * kE * 2);
    unsigned short* rb1w1_bf = (unsigned short*)alloc((size_t)kE * kE * 2);
    unsigned short* rb1w2_bf = (unsigned short*)alloc((size_t)kE * kE * 2);
    unsigned short* bsw_bf   = (unsigned short*)alloc((size_t)kE * kE * 2);
    unsigned short* rb2w1_bf = (unsigned short*)alloc((size_t)kE * kE * 2);
    unsigned short* rb2w2_bf = (unsigned short*)alloc((size_t)kE * kE * 2);
    unsigned short* rb3w1_bf = (unsigned short*)alloc((size_t)kE * kE * 2);
    unsigned short* rb3w2_bf = (unsigned short*)alloc((size_t)kE * kE * 2);
    if (off > ws_size) return;  // workspace too small; nothing sane to do

    auto CVT = [&](const float* s, unsigned short* d, size_t n) {
        cvt_f32_bf16_kernel<<<(unsigned)((n + 255) / 256), 256, 0, stream>>>(s, d, (int)n);
    };
    CVT(x, x_bf, (size_t)kNE * kE);
    CVT(rbf, rbf_bf, (size_t)kNE * kNR);
    CVT(w_rbf, w_rbf_bf, (size_t)kE * kNR);
    CVT(w_kj, w_kj_bf, (size_t)kE * kE);
    CVT(w_ji, w_ji_bf, (size_t)kE * kE);
    CVT(w_bil, wbil_bf, (size_t)kE * kNB * kE);
    CVT(rb1w1, rb1w1_bf, (size_t)kE * kE);
    CVT(rb1w2, rb1w2_bf, (size_t)kE * kE);
    CVT(bsw, bsw_bf, (size_t)kE * kE);
    CVT(rb2w1, rb2w1_bf, (size_t)kE * kE);
    CVT(rb2w2, rb2w2_bf, (size_t)kE * kE);
    CVT(rb3w1, rb3w1_bf, (size_t)kE * kE);
    CVT(rb3w2, rb3w2_bf, (size_t)kE * kE);

    edge_proj_kernel<<<(kEdgeTiles + 7) / 8, 256, 0, stream>>>(
        x_bf, rbf_bf, w_ji_bf, b_ji, w_kj_bf, b_kj, w_rbf_bf, agg, xkj_bf);

    sbf_proj_kernel<<<(kNT + 255) / 256, 256, 0, stream>>>(sbf, w_sbf, sbf_p);

    const int smem3 = 128 * 512 * 2 + 8 * 16 * kE * 2;  // 160 KB
    hipFuncSetAttribute(reinterpret_cast<const void*>(bilinear_kernel),
                        hipFuncAttributeMaxDynamicSharedMemorySize, smem3);
    dim3 g3((kTriTiles + 7) / 8, 2, 1);
    bilinear_kernel<<<g3, 256, smem3, stream>>>(xkj_bf, wbil_bf, sbf_p, idx_kj, idx_ji, agg);

    const int smem4 = 8 * 16 * kE * 2 + 8 * 16 * kE * 4;  // 96 KB
    hipFuncSetAttribute(reinterpret_cast<const void*>(epilogue_kernel),
                        hipFuncAttributeMaxDynamicSharedMemorySize, smem4);
    epilogue_kernel<<<(kEdgeTiles + 7) / 8, 256, smem4, stream>>>(
        agg, x, rb1w1_bf, rb1b1, rb1w2_bf, rb1b2, bsw_bf, bsb,
        rb2w1_bf, rb2b1, rb2w2_bf, rb2b2, rb3w1_bf, rb3b1, rb3w2_bf, rb3b2,
        (float*)d_out);
}
